// DETM_86071144612032
// MI455X (gfx1250) — compile-verified, matrix operand order checked by reading, not run on hardware
//
#include <hip/hip_runtime.h>
#include <math.h>

// ---------------------------------------------------------------------------
// DETM forward for MI455X (gfx1250, wave32).
// All GEMMs use V_WMMA_F32_16X16X32_BF16 (fp32 accumulate). The (T,K,V) beta
// tensor (200MB) is never materialized: softmax stats and the likelihood are
// computed in three streaming WMMA passes (online softmax), keeping the whole
// working set (~100MB) inside the 192MB L2.
// ---------------------------------------------------------------------------

#define DEVI __device__ __forceinline__

typedef __attribute__((ext_vector_type(16))) __bf16        v16bf;
typedef __attribute__((ext_vector_type(8)))  float         v8f;
typedef __attribute__((ext_vector_type(4)))  unsigned int  u32x4;

constexpr int V    = 20000, K = 50, T = 50, E = 300;
constexpr int TH   = 800, EH = 200, NL = 3, B = 64;
constexpr int EP   = 320;      // E padded to multiple of 32 (WMMA K-depth)
constexpr int MR   = 64;       // K topics padded to 64 rows (4 M-tiles)
constexpr int VPK  = 20064;    // V+K padded to multiple of 32
constexpr int EHP  = 208;      // EH padded to multiple of 16 (N-tiles)
constexpr int NT_V = V / 16;   // 1250 column tiles over vocab

// ---- helpers --------------------------------------------------------------

DEVI unsigned short f2bf(float x) {          // round-to-nearest-even f32->bf16
  unsigned int u = __float_as_uint(x);
  unsigned int r = u + 0x7FFFu + ((u >> 16) & 1u);
  return (unsigned short)(r >> 16);
}

// Fragment loader matching the ISA 16-bit A/B layout:
//   lane L: row = L&15, hi = L>>4; elems 0..7 = K[hi*8 .. hi*8+7],
//   elems 8..15 = K[16+hi*8 .. 16+hi*8+7]  -> two contiguous 16B chunks.
DEVI v16bf ldfrag(const unsigned short* p) {
  union { u32x4 u[2]; v16bf v; } r;
  r.u[0] = *(const u32x4*)(p);
  r.u[1] = *(const u32x4*)(p + 16);
  return r.v;
}

DEVI void atomicMaxF(float* addr, float val) {
  unsigned int* ua = (unsigned int*)addr;
  unsigned int old = *ua;
  while (true) {
    float cur = __uint_as_float(old);
    if (cur >= val) break;
    unsigned int assumed = old;
    old = atomicCAS(ua, assumed, __float_as_uint(val));
    if (old == assumed) break;
  }
}

DEVI float sigf(float x) { return 1.0f / (1.0f + expf(-x)); }

// ---- init / conversion ----------------------------------------------------

__global__ void k_init(float* scal, float* maxbuf, float* sumbuf) {
  int i = blockIdx.x * blockDim.x + threadIdx.x;
  if (i < 8) scal[i] = 0.0f;
  if (i < T * MR) { maxbuf[i] = -3.0e38f; sumbuf[i] = 0.0f; }
}

__global__ void k_convert_pad(const float* __restrict__ src,
                              unsigned short* __restrict__ dst,
                              int R, int C, int Rp, int Cp) {
  long i = (long)blockIdx.x * blockDim.x + threadIdx.x;
  long n = (long)Rp * Cp;
  if (i >= n) return;
  int r = (int)(i / Cp), c = (int)(i % Cp);
  float v = (r < R && c < C) ? src[(size_t)r * C + c] : 0.0f;
  dst[i] = f2bf(v);
}

// ---- alphas + kl_alpha ----------------------------------------------------

__global__ void k_alphas(const float* __restrict__ mu_q,
                         const float* __restrict__ ls_q,
                         const float* __restrict__ eps,
                         float* __restrict__ alphas) {
  long i = (long)blockIdx.x * blockDim.x + threadIdx.x;
  if (i >= (long)T * K * E) return;
  int t = (int)(i / (K * E));
  int r = (int)(i % (K * E));
  int k = r / E, e = r % E;
  size_t qi = ((size_t)k * T + t) * E + e;          // (K,T,E) -> transpose
  alphas[i] = mu_q[qi] + eps[i] * expf(0.5f * ls_q[qi]);
}

__global__ void k_alphas_bf(const float* __restrict__ alphas,
                            unsigned short* __restrict__ dst) {
  long i = (long)blockIdx.x * blockDim.x + threadIdx.x;
  if (i >= (long)T * MR * EP) return;
  int t = (int)(i / (MR * EP));
  int r = (int)(i % (MR * EP));
  int k = r / EP, e = r % EP;
  float v = (k < K && e < E) ? alphas[((size_t)t * K + k) * E + e] : 0.0f;
  dst[i] = f2bf(v);
}

__global__ __launch_bounds__(256)
void k_kl_alpha(const float* __restrict__ mu_q, const float* __restrict__ ls_q,
                const float* __restrict__ alphas, float* scal) {
  int tk = blockIdx.x;
  int t = tk / K, k = tk % K;
  const float logd = logf(0.005f);
  float part = 0.0f;
  for (int e = threadIdx.x; e < E; e += blockDim.x) {
    size_t qi = ((size_t)k * T + t) * E + e;
    float mu = mu_q[qi], ls = ls_q[qi];
    if (t == 0) {
      part += 0.5f * ((expf(ls) + mu * mu) / (1.0f + 1e-6f) - 1.0f - ls);
    } else {
      float pm = alphas[((size_t)(t - 1) * K + k) * E + e];
      float d = mu - pm;
      part += 0.5f * ((expf(ls) + d * d) / (0.005f + 1e-6f) - 1.0f + logd - ls);
    }
  }
  __shared__ float red[256];
  red[threadIdx.x] = part;
  __syncthreads();
  for (int s = 128; s; s >>= 1) {
    if (threadIdx.x < s) red[threadIdx.x] += red[threadIdx.x + s];
    __syncthreads();
  }
  if (threadIdx.x == 0) atomicAdd(&scal[0], red[0]);
}

// ---- generic bf16 WMMA GEMM: C[M,N] = A(M,Kp) * B(N,Kp)^T ----------------
// A, B stored row-major with K contiguous; one 16x16 C tile per wave.

__global__ __launch_bounds__(256)
void k_gemm(const unsigned short* __restrict__ A,
            const unsigned short* __restrict__ Bm,
            float* __restrict__ C,
            int Ntiles, int Ksteps, int lda, int ldb, int ldc) {
  const int lane = threadIdx.x & 31;
  const int wave = threadIdx.x >> 5;
  const int nt = blockIdx.x * 8 + wave;
  const int mt = blockIdx.y;
  if (nt >= Ntiles) return;
  const int rlo = lane & 15, hi = lane >> 4;
  const unsigned short* arow = A + (size_t)(mt * 16 + rlo) * lda;
  const unsigned short* brow = Bm + (size_t)(nt * 16 + rlo) * ldb;
  v8f acc = {};
  for (int ks = 0; ks < Ksteps; ++ks) {
    int kk = ks * 32 + hi * 8;
    v16bf a = ldfrag(arow + kk);
    v16bf b = ldfrag(brow + kk);
    acc = __builtin_amdgcn_wmma_f32_16x16x32_bf16(false, a, false, b,
                                                  (short)0, acc, false, false);
  }
  const int m0 = mt * 16 + hi * 8;
  const int n  = nt * 16 + rlo;
#pragma unroll
  for (int r = 0; r < 8; ++r) C[(size_t)(m0 + r) * ldc + n] = acc[r];
}

// ---- LSTM (3 layers, batch 1, T=50) + eta scan, single workgroup ---------

__global__ __launch_bounds__(1024)
void k_lstm_eta(const float* __restrict__ x0pre, const float* __restrict__ emb,
                const float* __restrict__ Wih, const float* __restrict__ Whh,
                const float* __restrict__ bih, const float* __restrict__ bhh,
                const float* __restrict__ meW, const float* __restrict__ meb,
                const float* __restrict__ leW, const float* __restrict__ leb,
                const float* __restrict__ eps_eta,
                float* __restrict__ etas_g, float* scal) {
  __shared__ float xbuf[T * EH];         // whole sequence lives in LDS (40KB)
  __shared__ float hbuf[EH], cbuf[EH], gates[4 * EH];
  __shared__ float eta_prev[K], red[K];
  const int tid = threadIdx.x;

  for (int i = tid; i < T * EH; i += blockDim.x) {
    int t = i / EH, e = i % EH;
    xbuf[i] = x0pre[(size_t)t * EHP + e] + emb[e];
  }
  __syncthreads();

  for (int l = 0; l < NL; ++l) {
    for (int i = tid; i < EH; i += blockDim.x) { hbuf[i] = 0.0f; cbuf[i] = 0.0f; }
    __syncthreads();
    const float* wi0 = Wih + (size_t)l * 4 * EH * EH;
    const float* wh0 = Whh + (size_t)l * 4 * EH * EH;
    for (int t = 0; t < T; ++t) {
      if (tid < 4 * EH) {
        float g = bih[l * 4 * EH + tid] + bhh[l * 4 * EH + tid];
        const float* wi = wi0 + (size_t)tid * EH;
        const float* wh = wh0 + (size_t)tid * EH;
        const float* xt = &xbuf[t * EH];
        for (int e = 0; e < EH; ++e) g += wi[e] * xt[e] + wh[e] * hbuf[e];
        gates[tid] = g;
      }
      __syncthreads();
      if (tid < EH) {
        float ig = sigf(gates[tid]);
        float fg = sigf(gates[EH + tid]);
        float gg = tanhf(gates[2 * EH + tid]);
        float og = sigf(gates[3 * EH + tid]);
        float c = fg * cbuf[tid] + ig * gg;
        cbuf[tid] = c;
        float h = og * tanhf(c);
        hbuf[tid] = h;
        xbuf[t * EH + tid] = h;            // layer output feeds next layer
      }
      __syncthreads();
    }
  }

  // eta scan (sequential chain over T)
  const float logd = logf(0.005f);
  float klacc = 0.0f, eta_new = 0.0f;
  if (tid < K) {
    float mu = meb[tid], ls = leb[tid];
    const float* mw = meW + (size_t)tid * (EH + K);
    const float* lw = leW + (size_t)tid * (EH + K);
    for (int j = 0; j < EH; ++j) { float xv = xbuf[j]; mu += mw[j] * xv; ls += lw[j] * xv; }
    eta_new = mu + eps_eta[tid] * expf(0.5f * ls);
    etas_g[tid] = eta_new;
    klacc += 0.5f * ((expf(ls) + mu * mu) / (1.0f + 1e-6f) - 1.0f - ls);
  }
  __syncthreads();
  if (tid < K) eta_prev[tid] = eta_new;
  __syncthreads();
  for (int t = 1; t < T; ++t) {
    if (tid < K) {
      float mu = meb[tid], ls = leb[tid];
      const float* mw = meW + (size_t)tid * (EH + K);
      const float* lw = leW + (size_t)tid * (EH + K);
      const float* xt = &xbuf[t * EH];
      for (int j = 0; j < EH; ++j) { float xv = xt[j]; mu += mw[j] * xv; ls += lw[j] * xv; }
      for (int j = 0; j < K; ++j)  { float ev = eta_prev[j]; mu += mw[EH + j] * ev; ls += lw[EH + j] * ev; }
      eta_new = mu + eps_eta[t * K + tid] * expf(0.5f * ls);
      etas_g[t * K + tid] = eta_new;
      float d = mu - eta_prev[tid];
      klacc += 0.5f * ((expf(ls) + d * d) / (0.005f + 1e-6f) - 1.0f + logd - ls);
    }
    __syncthreads();
    if (tid < K) eta_prev[tid] = eta_new;
    __syncthreads();
  }
  if (tid < K) red[tid] = klacc;
  __syncthreads();
  if (tid == 0) {
    float s = 0.0f;
    for (int j = 0; j < K; ++j) s += red[j];
    scal[1] = s;
  }
}

// ---- encoder input staging + relu/bf16 ------------------------------------

__global__ void k_build_inp(const float* __restrict__ nb,
                            const float* __restrict__ etas,
                            const int* __restrict__ times,
                            unsigned short* __restrict__ dst) {
  long i = (long)blockIdx.x * blockDim.x + threadIdx.x;
  if (i >= (long)B * VPK) return;
  int b = (int)(i / VPK), c = (int)(i % VPK);
  float v;
  if (c < V)            v = nb[(size_t)b * V + c];
  else if (c < V + K)   v = etas[(size_t)times[b] * K + (c - V)];
  else                  v = 0.0f;
  dst[i] = f2bf(v);
}

__global__ void k_relu_bf(const float* __restrict__ pre,
                          const float* __restrict__ bias,
                          unsigned short* __restrict__ dst) {
  long i = (long)blockIdx.x * blockDim.x + threadIdx.x;
  if (i >= (long)B * TH) return;
  int c = (int)(i % TH);
  dst[i] = f2bf(fmaxf(pre[i] + bias[c], 0.0f));
}

// ---- theta head: relu, mu/ls, softmax, kl_theta ---------------------------

__global__ __launch_bounds__(256)
void k_theta(const float* __restrict__ h2pre, const float* __restrict__ b2,
             const float* __restrict__ muW, const float* __restrict__ mub,
             const float* __restrict__ lsW, const float* __restrict__ lsb,
             const float* __restrict__ eps_theta, const float* __restrict__ etas,
             const int* __restrict__ times, float* __restrict__ theta_b,
             float* scal) {
  __shared__ float h2[TH];
  __shared__ float zb[K], mus[K], lss[K];
  __shared__ float mx, sm;
  const int b = blockIdx.x, tid = threadIdx.x;
  for (int j = tid; j < TH; j += blockDim.x)
    h2[j] = fmaxf(h2pre[(size_t)b * TH + j] + b2[j], 0.0f);
  __syncthreads();
  if (tid < K) {
    float mu = mub[tid], ls = lsb[tid];
    const float* mw = muW + (size_t)tid * TH;
    const float* lw = lsW + (size_t)tid * TH;
    for (int j = 0; j < TH; ++j) { float h = h2[j]; mu += mw[j] * h; ls += lw[j] * h; }
    mus[tid] = mu; lss[tid] = ls;
    zb[tid] = mu + eps_theta[(size_t)b * K + tid] * expf(0.5f * ls);
  }
  __syncthreads();
  if (tid == 0) {
    float m = zb[0];
    for (int j = 1; j < K; ++j) m = fmaxf(m, zb[j]);
    float s = 0.0f;
    for (int j = 0; j < K; ++j) s += expf(zb[j] - m);
    mx = m; sm = s;
  }
  __syncthreads();
  if (tid < MR) theta_b[(size_t)b * MR + tid] = (tid < K) ? expf(zb[tid] - mx) / sm : 0.0f;
  if (tid < K) {
    float etd = etas[(size_t)times[b] * K + tid];
    float d = mus[tid] - etd;
    zb[tid] = 0.5f * ((expf(lss[tid]) + d * d) / (1.0f + 1e-6f) - 1.0f - lss[tid]);
  }
  __syncthreads();
  if (tid == 0) {
    float s = 0.0f;
    for (int j = 0; j < K; ++j) s += zb[j];
    atomicAdd(&scal[2], s);
  }
}

// ---- beta passes: streaming WMMA over alphas[t] x rho^T -------------------
// Per wave: 4 M-tiles (64 padded topic rows) x one 16-col vocab tile.

DEVI void beta_tile(const unsigned short* __restrict__ Abase,
                    const unsigned short* __restrict__ Brow,
                    int rlo, int hi, v8f acc[4]) {
  for (int ks = 0; ks < EP / 32; ++ks) {
    int kk = ks * 32 + hi * 8;
    v16bf b = ldfrag(Brow + kk);
#pragma unroll
    for (int mt = 0; mt < 4; ++mt) {
      v16bf a = ldfrag(Abase + (size_t)(mt * 16 + rlo) * EP + kk);
      acc[mt] = __builtin_amdgcn_wmma_f32_16x16x32_bf16(false, a, false, b,
                                                        (short)0, acc[mt], false, false);
    }
  }
}

__global__ __launch_bounds__(256)
void k_beta_max(const unsigned short* __restrict__ alphas_bf,
                const unsigned short* __restrict__ rho_bf,
                float* __restrict__ maxbuf) {
  const int lane = threadIdx.x & 31, wave = threadIdx.x >> 5;
  const int nt = blockIdx.x * 8 + wave, t = blockIdx.y;
  if (nt >= NT_V) return;
  const int rlo = lane & 15, hi = lane >> 4;
  v8f acc[4] = {{}, {}, {}, {}};
  beta_tile(alphas_bf + (size_t)t * MR * EP,
            rho_bf + (size_t)(nt * 16 + rlo) * EP, rlo, hi, acc);
#pragma unroll
  for (int mt = 0; mt < 4; ++mt)
#pragma unroll
    for (int r = 0; r < 8; ++r) {
      float v = acc[mt][r];
      v = fmaxf(v, __shfl_xor(v, 1, 32));
      v = fmaxf(v, __shfl_xor(v, 2, 32));
      v = fmaxf(v, __shfl_xor(v, 4, 32));
      v = fmaxf(v, __shfl_xor(v, 8, 32));
      if (rlo == 0) atomicMaxF(&maxbuf[t * MR + mt * 16 + hi * 8 + r], v);
    }
}

__global__ __launch_bounds__(256)
void k_beta_sum(const unsigned short* __restrict__ alphas_bf,
                const unsigned short* __restrict__ rho_bf,
                const float* __restrict__ maxbuf,
                float* __restrict__ sumbuf) {
  const int lane = threadIdx.x & 31, wave = threadIdx.x >> 5;
  const int nt = blockIdx.x * 8 + wave, t = blockIdx.y;
  if (nt >= NT_V) return;
  const int rlo = lane & 15, hi = lane >> 4;
  v8f acc[4] = {{}, {}, {}, {}};
  beta_tile(alphas_bf + (size_t)t * MR * EP,
            rho_bf + (size_t)(nt * 16 + rlo) * EP, rlo, hi, acc);
#pragma unroll
  for (int mt = 0; mt < 4; ++mt)
#pragma unroll
    for (int r = 0; r < 8; ++r) {
      int m = mt * 16 + hi * 8 + r;
      float s = expf(acc[mt][r] - maxbuf[t * MR + m]);
      s += __shfl_xor(s, 1, 32);
      s += __shfl_xor(s, 2, 32);
      s += __shfl_xor(s, 4, 32);
      s += __shfl_xor(s, 8, 32);
      if (rlo == 0) atomicAdd(&sumbuf[t * MR + m], s);
    }
}

__global__ __launch_bounds__(256)
void k_beta_lik(const unsigned short* __restrict__ alphas_bf,
                const unsigned short* __restrict__ rho_bf,
                const float* __restrict__ maxbuf, const float* __restrict__ sumbuf,
                const float* __restrict__ theta_b, const float* __restrict__ nb,
                const int* __restrict__ times, float* scal) {
  const int lane = threadIdx.x & 31, wave = threadIdx.x >> 5;
  const int nt = blockIdx.x * 8 + wave, b = blockIdx.y;
  if (nt >= NT_V) return;
  const int t = times[b];
  const int rlo = lane & 15, hi = lane >> 4;
  v8f acc[4] = {{}, {}, {}, {}};
  beta_tile(alphas_bf + (size_t)t * MR * EP,
            rho_bf + (size_t)(nt * 16 + rlo) * EP, rlo, hi, acc);
  float lik = 0.0f;
#pragma unroll
  for (int mt = 0; mt < 4; ++mt)
#pragma unroll
    for (int r = 0; r < 8; ++r) {
      int m = mt * 16 + hi * 8 + r;
      float th = theta_b[(size_t)b * MR + m];   // zero for padded rows m>=50
      lik += th * expf(acc[mt][r] - maxbuf[t * MR + m]) / sumbuf[t * MR + m];
    }
  lik += __shfl_xor(lik, 16, 32);               // combine row halves (same col)
  float contrib = 0.0f;
  if (hi == 0) {
    int v = nt * 16 + rlo;
    contrib = -logf(lik + 1e-6f) * nb[(size_t)b * V + v];
  }
  for (int off = 16; off; off >>= 1) contrib += __shfl_xor(contrib, off, 32);
  if (lane == 0) atomicAdd(&scal[3], contrib);
}

__global__ void k_final(const float* scal, float* out) {
  if (threadIdx.x == 0 && blockIdx.x == 0) {
    float nll = scal[3] / (float)B;
    float kla = scal[0], kle = scal[1], klt = scal[2] / (float)B;
    out[0] = nll + kla + kle + klt;
    out[1] = nll; out[2] = kla; out[3] = kle; out[4] = klt;
  }
}

// ---------------------------------------------------------------------------

extern "C" void kernel_launch(void* const* d_in, const int* in_sizes, int n_in,
                              void* d_out, int out_size, void* d_ws, size_t ws_size,
                              hipStream_t stream) {
  (void)in_sizes; (void)n_in; (void)out_size; (void)ws_size;
  const float* nb        = (const float*)d_in[1];
  const int*   times     = (const int*)  d_in[2];
  const float* rnn_inp   = (const float*)d_in[3];
  const float* eps_alpha = (const float*)d_in[5];
  const float* eps_eta   = (const float*)d_in[6];
  const float* eps_theta = (const float*)d_in[7];
  const float* rho       = (const float*)d_in[8];
  const float* mu_q_a    = (const float*)d_in[9];
  const float* ls_q_a    = (const float*)d_in[10];
  const float* W1  = (const float*)d_in[11];
  const float* b1  = (const float*)d_in[12];
  const float* W2  = (const float*)d_in[13];
  const float* b2  = (const float*)d_in[14];
  const float* muW = (const float*)d_in[15];
  const float* mub = (const float*)d_in[16];
  const float* lsW = (const float*)d_in[17];
  const float* lsb = (const float*)d_in[18];
  const float* emW = (const float*)d_in[19];
  const float* emb = (const float*)d_in[20];
  const float* Wih = (const float*)d_in[21];
  const float* Whh = (const float*)d_in[22];
  const float* bih = (const float*)d_in[23];
  const float* bhh = (const float*)d_in[24];
  const float* meW = (const float*)d_in[25];
  const float* meb = (const float*)d_in[26];
  const float* leW = (const float*)d_in[27];
  const float* leb = (const float*)d_in[28];

  // bump allocator over d_ws (256B aligned so all b128 fragment loads align)
  char* base = (char*)d_ws; size_t off = 0;
  auto alloc = [&](size_t bytes) -> void* {
    void* p = base + off;
    off = (off + bytes + 255) & ~(size_t)255;
    return p;
  };
  float* scal   = (float*)alloc(8 * sizeof(float));
  float* maxbuf = (float*)alloc((size_t)T * MR * sizeof(float));
  float* sumbuf = (float*)alloc((size_t)T * MR * sizeof(float));
  float* alphas = (float*)alloc((size_t)T * K * E * sizeof(float));
  unsigned short* alphas_bf = (unsigned short*)alloc((size_t)T * MR * EP * 2);
  unsigned short* rho_bf    = (unsigned short*)alloc((size_t)V * EP * 2);
  unsigned short* rnn_bf    = (unsigned short*)alloc((size_t)MR * V * 2);
  unsigned short* emW_bf    = (unsigned short*)alloc((size_t)EHP * V * 2);
  float* x0pre  = (float*)alloc((size_t)MR * EHP * sizeof(float));
  float* etas   = (float*)alloc((size_t)T * K * sizeof(float));
  unsigned short* inp_bf = (unsigned short*)alloc((size_t)B * VPK * 2);
  unsigned short* W1_bf  = (unsigned short*)alloc((size_t)TH * VPK * 2);
  float* h1pre  = (float*)alloc((size_t)B * TH * sizeof(float));
  unsigned short* h1_bf = (unsigned short*)alloc((size_t)B * TH * 2);
  unsigned short* W2_bf = (unsigned short*)alloc((size_t)TH * TH * 2);
  float* h2pre  = (float*)alloc((size_t)B * TH * sizeof(float));
  float* theta_b = (float*)alloc((size_t)B * MR * sizeof(float));

  auto g1 = [](long n) { return dim3((unsigned)((n + 255) / 256)); };

  k_init<<<g1(T * MR), 256, 0, stream>>>(scal, maxbuf, sumbuf);

  k_convert_pad<<<g1((long)V * EP),  256, 0, stream>>>(rho,     rho_bf, V,  E,     V,   EP);
  k_convert_pad<<<g1((long)MR * V),  256, 0, stream>>>(rnn_inp, rnn_bf, T,  V,     MR,  V);
  k_convert_pad<<<g1((long)EHP * V), 256, 0, stream>>>(emW,     emW_bf, EH, V,     EHP, V);
  k_convert_pad<<<g1((long)TH * VPK),256, 0, stream>>>(W1,      W1_bf,  TH, V + K, TH,  VPK);
  k_convert_pad<<<g1((long)TH * TH), 256, 0, stream>>>(W2,      W2_bf,  TH, TH,    TH,  TH);

  k_alphas<<<g1((long)T * K * E),  256, 0, stream>>>(mu_q_a, ls_q_a, eps_alpha, alphas);
  k_alphas_bf<<<g1((long)T * MR * EP), 256, 0, stream>>>(alphas, alphas_bf);
  k_kl_alpha<<<dim3(T * K), 256, 0, stream>>>(mu_q_a, ls_q_a, alphas, scal);

  // x0 = rnn_inp @ eta_map_W^T : M=64(pad), N=208(pad), Ksteps=625
  k_gemm<<<dim3(2, 4), 256, 0, stream>>>(rnn_bf, emW_bf, x0pre, EHP / 16, V / 32, V, V, EHP);
  k_lstm_eta<<<dim3(1), 1024, 0, stream>>>(x0pre, emb, Wih, Whh, bih, bhh,
                                           meW, meb, leW, leb, eps_eta, etas, scal);

  k_build_inp<<<g1((long)B * VPK), 256, 0, stream>>>(nb, etas, times, inp_bf);
  // h1pre = inp @ W1^T : M=64, N=800, Ksteps=627
  k_gemm<<<dim3(7, 4), 256, 0, stream>>>(inp_bf, W1_bf, h1pre, TH / 16, VPK / 32, VPK, VPK, TH);
  k_relu_bf<<<g1((long)B * TH), 256, 0, stream>>>(h1pre, b1, h1_bf);
  // h2pre = h1 @ W2^T : M=64, N=800, Ksteps=25
  k_gemm<<<dim3(7, 4), 256, 0, stream>>>(h1_bf, W2_bf, h2pre, TH / 16, TH / 32, TH, TH, TH);
  k_theta<<<dim3(B), 256, 0, stream>>>(h2pre, b2, muW, mub, lsW, lsb,
                                       eps_theta, etas, times, theta_b, scal);

  // online-softmax beta passes (no 200MB beta tensor materialized)
  k_beta_max<<<dim3((NT_V + 7) / 8, T), 256, 0, stream>>>(alphas_bf, rho_bf, maxbuf);
  k_beta_sum<<<dim3((NT_V + 7) / 8, T), 256, 0, stream>>>(alphas_bf, rho_bf, maxbuf, sumbuf);
  k_beta_lik<<<dim3((NT_V + 7) / 8, B), 256, 0, stream>>>(alphas_bf, rho_bf, maxbuf, sumbuf,
                                                          theta_b, nb, times, scal);

  k_final<<<dim3(1), 1, 0, stream>>>(scal, (float*)d_out);
}
